// GraphMambaLayer_18837726560374
// MI455X (gfx1250) — matrix operand order repaired and checked by reading, not compiled, not run on hardware
//
#include <hip/hip_runtime.h>
#include <hip/hip_bf16.h>
#include <math.h>

// ---------------------------------------------------------------------------
// GraphMamba layer for MI455X (gfx1250), compile-only target.
//
// Exact simplification: the seq models are per-token MLPs, so the ordering /
// reverse / unscramble permutations cancel exactly.  The layer is a per-token
// GEMM chain, run as fused WMMA bf16 GEMMs (fp32 accumulate) with:
//   - Tensor Data Mover staging of 64x256 weight slices into LDS (double
//     buffered: tensor_load_to_lds + s_wait_tensorcnt),
//   - A fragments streamed from global (b128 clauses + prefetch),
//   - bias / SiLU / GELU / LayerNorm / residual / mask fused in epilogues.
// ---------------------------------------------------------------------------

typedef __bf16 bf16;
typedef __attribute__((ext_vector_type(16))) __bf16 v16bf;
typedef __attribute__((ext_vector_type(8)))  __bf16 v8bf;   // 16-byte piece
typedef __attribute__((ext_vector_type(8)))  float  v8f;
typedef __attribute__((ext_vector_type(4)))  unsigned int v4u;
typedef __attribute__((ext_vector_type(8)))  unsigned int v8u;

#define MTOK   (2048 * 64)   // B*K tokens
#define DD     256
#define DINNER 512
#define DFF    1024
#define NLAYER 2
#define KSLICE 64            // K-depth staged per TDM transfer

enum EpiKind { EPI_SILU = 0, EPI_GELU = 1, EPI_LN = 2, EPI_MERGE = 3, EPI_OUT = 4 };

__device__ __forceinline__ float siluf(float x) { return x / (1.0f + __expf(-x)); }
__device__ __forceinline__ float geluf(float x) {
  return 0.5f * x * (1.0f + erff(x * 0.70710678118654752f));
}

__global__ __launch_bounds__(256) void cvt_f32_bf16(const float* __restrict__ s,
                                                    bf16* __restrict__ d, int n) {
  int i = blockIdx.x * 256 + threadIdx.x;
  if (i < n) d[i] = (bf16)s[i];
}

// ---------------------------------------------------------------------------
// TDM: DMA one [rows x cols] bf16 tile (row stride ldb elements) from global
// into LDS at byte offset lds_off.  2-D descriptor (D# groups 0+1), no
// gather / iterate / pad / multicast.  Issued once per calling wave; tracked
// with TENSORcnt.  Inline asm per CDNA5_HIP.md (portable across toolchains).
// ---------------------------------------------------------------------------
__device__ __forceinline__ void tdm_load_tile_2d(const bf16* gsrc, unsigned ldb,
                                                 unsigned lds_off,
                                                 unsigned rows, unsigned cols) {
  unsigned long long ga = (unsigned long long)(uintptr_t)gsrc;
  v4u g0;
  g0[0] = 1u;                                   // count=1, user mode, no gather
  g0[1] = lds_off;                              // lds_addr (bytes)
  g0[2] = (unsigned)(ga & 0xffffffffu);         // global_addr[31:0]
  g0[3] = (unsigned)((ga >> 32) & 0x01ffffffu)  // global_addr[56:32]
          | (2u << 30);                         // type = 2 ("image")
  v8u g1;
  g1[0] = (1u << 16);                           // data_size=1 (2B); mask=0
  g1[1] = (ldb & 0xffffu) << 16;                // tensor_dim0[15:0]
  g1[2] = (ldb >> 16) | ((rows & 0xffffu) << 16);  // td0[31:16] | tensor_dim1 lo
  g1[3] = (cols & 0xffffu) << 16;               // tensor_dim1 hi(=0) | tile_dim0
  g1[4] = rows & 0xffffu;                       // tile_dim1 | tile_dim2=0
  g1[5] = ldb;                                  // tensor_dim0_stride[31:0]
  g1[6] = 0u;                                   // stride hi | dim1_stride lo
  g1[7] = 0u;                                   // dim1_stride hi (unused, 2-D)
  asm volatile("tensor_load_to_lds %0, %1" : : "s"(g0), "s"(g1) : "memory");
}

// Row-wise mean/rstd over the block's 256 columns (32 rows per block).
__device__ __forceinline__ void row_stats_256(float* s_sum, float* s_ssq,
                                              const float (&x)[4][8], int rb,
                                              float (&mu)[8], float (&rs)[8]) {
  const int tid = threadIdx.x;
  if (tid < 32) { s_sum[tid] = 0.0f; s_ssq[tid] = 0.0f; }
  __syncthreads();
#pragma unroll
  for (int r = 0; r < 8; ++r) {
    float ps = x[0][r] + x[1][r] + x[2][r] + x[3][r];
    float pq = x[0][r] * x[0][r] + x[1][r] * x[1][r] +
               x[2][r] * x[2][r] + x[3][r] * x[3][r];
#pragma unroll
    for (int m = 8; m >= 1; m >>= 1) {
      ps += __shfl_xor(ps, m, 32);
      pq += __shfl_xor(pq, m, 32);
    }
    if ((tid & 15) == 0) {
      atomicAdd(&s_sum[rb + r], ps);
      atomicAdd(&s_ssq[rb + r], pq);
    }
  }
  __syncthreads();
#pragma unroll
  for (int r = 0; r < 8; ++r) {
    float m_ = s_sum[rb + r] * (1.0f / 256.0f);
    float v_ = s_ssq[rb + r] * (1.0f / 256.0f) - m_ * m_;
    mu[r] = m_;
    rs[r] = rsqrtf(v_ + 1e-5f);
  }
  __syncthreads();
}

// Generic fused GEMM:  out = Epi( A[M x K](bf16) @ Bw[K x Ntot](bf16) + bias )
// Block tile 32x256, 8 waves; WMMA 16x16x32 bf16 with fp32 accum.
// B is staged KSLICE-deep per TDM transfer into LDS (double buffered);
// 8 WMMAs + 4 A-loads per barrier pair.
template <int EPI>
__global__ __launch_bounds__(256, 2) void wmma_mlp(
    const bf16* __restrict__ A, int lda,
    const bf16* __restrict__ Bw, int ldb,
    const float* __restrict__ bias, int K,
    bf16* __restrict__ outb, int ldob,
    const float* __restrict__ lnw, const float* __restrict__ lnb,
    const float* __restrict__ ln2w, const float* __restrict__ ln2b,
    const float* __restrict__ resid, int ldr,
    float* __restrict__ outf, int ldof,
    const float* __restrict__ keep) {
  __shared__ __align__(16) bf16 bstage[2][KSLICE * 256];  // 2 x 32KB B slices
  __shared__ float s_sum[32];
  __shared__ float s_ssq[32];

  const int tid  = threadIdx.x;
  const int wave = tid >> 5;
  const int lane = tid & 31;
  const int wm   = wave >> 2;  // 0..1: row subtile
  const int wnq  = wave & 3;   // 0..3: group of 4 col subtiles
  const int hlf  = lane >> 4;  // lane half (ISA A/C layout split)
  const int l16  = lane & 15;

  const int m0    = blockIdx.x * 32 + wm * 16;
  const int nbase = blockIdx.y * 256;
  const int nw0   = nbase + wnq * 64;

  v8f zero = {0.f, 0.f, 0.f, 0.f, 0.f, 0.f, 0.f, 0.f};
  v8f acc[4];
  acc[0] = zero; acc[1] = zero; acc[2] = zero; acc[3] = zero;

  // A fragment (16x32 bf16, ISA 7.12.2): lane = M row; lane-half selects the
  // two 8-wide K runs: half*8..+8 and 16+half*8..+8.
  const bf16* arow = A + (size_t)(m0 + l16) * lda;

  const int nslice = K / KSLICE;
  // Prologue: wave 0 kicks off the DMA of slice 0.
  if (wave == 0)
    tdm_load_tile_2d(Bw + nbase, (unsigned)ldb,
                     (unsigned)(uintptr_t)&bstage[0][0], KSLICE, 256u);

  for (int s = 0; s < nslice; ++s) {
    const int cur = s & 1;
    if (wave == 0) {
      if (s + 1 < nslice) {
        // Issue next slice into the other buffer, then wait for current one.
        tdm_load_tile_2d(Bw + (size_t)(s + 1) * KSLICE * ldb + nbase,
                         (unsigned)ldb,
                         (unsigned)(uintptr_t)&bstage[cur ^ 1][0], KSLICE, 256u);
        __builtin_amdgcn_s_wait_tensorcnt(1);  // in-order: slice s complete
      } else {
        __builtin_amdgcn_s_wait_tensorcnt(0);
      }
    }
    __syncthreads();  // B slice s visible to all waves

#pragma unroll
    for (int kk = 0; kk < KSLICE / 32; ++kk) {
      const int kb = s * KSLICE + kk * 32;
      v8bf a0 = *(const v8bf*)(arow + kb + hlf * 8);
      v8bf a1 = *(const v8bf*)(arow + kb + 16 + hlf * 8);
      v16bf av = __builtin_shufflevector(a0, a1, 0, 1, 2, 3, 4, 5, 6, 7,
                                         8, 9, 10, 11, 12, 13, 14, 15);
      __builtin_prefetch(arow + kb + KSLICE, 0, 0);
      // B fragment (32x16 bf16): lane = N column; K runs half*8..+8 and 16+half*8..+8 — from LDS.
      const bf16* bl = &bstage[cur][(kk * 32 + hlf * 8) * 256 + wnq * 64 + l16];
#pragma unroll
      for (int t = 0; t < 4; ++t) {
        v16bf bv;
#pragma unroll
        for (int i = 0; i < 8; ++i) {
          bv[i]     = bl[i * 256 + t * 16];
          bv[8 + i] = bl[(16 + i) * 256 + t * 16];
        }
        acc[t] = __builtin_amdgcn_wmma_f32_16x16x32_bf16(
            false, av, false, bv, (short)0, acc[t], false, false);
      }
    }
    __syncthreads();  // compute on slice s done before buffer reuse
  }

  // C layout: lane l -> N = n + (l&15); elem r -> M = m0 + (l>>4)*8 + r.
  int nt[4];
#pragma unroll
  for (int t = 0; t < 4; ++t) nt[t] = nw0 + t * 16 + l16;
  const int rbase = m0 + hlf * 8;        // global row for r=0
  const int rb    = wm * 16 + hlf * 8;   // row-in-block for r=0

  float x[4][8];
#pragma unroll
  for (int t = 0; t < 4; ++t) {
    const float bv = bias[nt[t]];
#pragma unroll
    for (int r = 0; r < 8; ++r) x[t][r] = acc[t][r] + bv;
  }

  if constexpr (EPI == EPI_SILU || EPI == EPI_GELU) {
#pragma unroll
    for (int t = 0; t < 4; ++t)
#pragma unroll
      for (int r = 0; r < 8; ++r) {
        float y = (EPI == EPI_SILU) ? siluf(x[t][r]) : geluf(x[t][r]);
        outb[(size_t)(rbase + r) * ldob + nt[t]] = (bf16)y;
      }
  } else if constexpr (EPI == EPI_LN) {
    float mu[8], rs[8];
    row_stats_256(s_sum, s_ssq, x, rb, mu, rs);
#pragma unroll
    for (int t = 0; t < 4; ++t) {
      float w = lnw[nt[t]], b = lnb[nt[t]];
#pragma unroll
      for (int r = 0; r < 8; ++r) {
        float y = (x[t][r] - mu[r]) * rs[r] * w + b;
        outb[(size_t)(rbase + r) * ldob + nt[t]] = (bf16)y;
      }
    }
  } else if constexpr (EPI == EPI_MERGE) {
    // x = merged + residual; slots = LN1(x) -> outf (f32, in-place safe);
    // h = LN2(slots) -> outb (bf16 FFN input)
#pragma unroll
    for (int t = 0; t < 4; ++t)
#pragma unroll
      for (int r = 0; r < 8; ++r)
        x[t][r] += resid[(size_t)(rbase + r) * ldr + nt[t]];
    float mu[8], rs[8];
    row_stats_256(s_sum, s_ssq, x, rb, mu, rs);
#pragma unroll
    for (int t = 0; t < 4; ++t) {
      float w = lnw[nt[t]], b = lnb[nt[t]];
#pragma unroll
      for (int r = 0; r < 8; ++r) {
        x[t][r] = (x[t][r] - mu[r]) * rs[r] * w + b;
        outf[(size_t)(rbase + r) * ldof + nt[t]] = x[t][r];
      }
    }
    float mu2[8], rs2[8];
    row_stats_256(s_sum, s_ssq, x, rb, mu2, rs2);
#pragma unroll
    for (int t = 0; t < 4; ++t) {
      float w = ln2w[nt[t]], b = ln2b[nt[t]];
#pragma unroll
      for (int r = 0; r < 8; ++r) {
        float y = (x[t][r] - mu2[r]) * rs2[r] * w + b;
        outb[(size_t)(rbase + r) * ldob + nt[t]] = (bf16)y;
      }
    }
  } else {  // EPI_OUT: y = (x + resid) * keep -> outf (f32) and outb (bf16)
#pragma unroll
    for (int r = 0; r < 8; ++r) {
      float km = keep[rbase + r];
#pragma unroll
      for (int t = 0; t < 4; ++t) {
        float y = (x[t][r] + resid[(size_t)(rbase + r) * ldr + nt[t]]) * km;
        outf[(size_t)(rbase + r) * ldof + nt[t]] = y;
        outb[(size_t)(rbase + r) * ldob + nt[t]] = (bf16)y;
      }
    }
  }
}

extern "C" void kernel_launch(void* const* d_in, const int* in_sizes, int n_in,
                              void* d_out, int out_size, void* d_ws, size_t ws_size,
                              hipStream_t stream) {
  (void)in_sizes; (void)n_in; (void)out_size;

  const float* slots   = (const float*)d_in[0];
  const float* keep    = (const float*)d_in[1];
  const float* fw_w1   = (const float*)d_in[2];
  const float* fw_b1   = (const float*)d_in[3];
  const float* fw_w2   = (const float*)d_in[4];
  const float* fw_b2   = (const float*)d_in[5];
  const float* fw_lnw  = (const float*)d_in[6];
  const float* fw_lnb  = (const float*)d_in[7];
  const float* bw_w1   = (const float*)d_in[8];
  const float* bw_b1   = (const float*)d_in[9];
  const float* bw_w2   = (const float*)d_in[10];
  const float* bw_b2   = (const float*)d_in[11];
  const float* bw_lnw  = (const float*)d_in[12];
  const float* bw_lnb  = (const float*)d_in[13];
  const float* mg_w    = (const float*)d_in[14];
  const float* mg_b    = (const float*)d_in[15];
  const float* nm_w    = (const float*)d_in[16];
  const float* nm_b    = (const float*)d_in[17];
  const float* ffn_lnw = (const float*)d_in[18];
  const float* ffn_lnb = (const float*)d_in[19];
  const float* ffn_w1  = (const float*)d_in[20];
  const float* ffn_b1  = (const float*)d_in[21];
  const float* ffn_w2  = (const float*)d_in[22];
  const float* ffn_b2  = (const float*)d_in[23];

  // --- workspace layout ---
  constexpr size_t S_OFF  = 0;                                   // slots f32
  constexpr size_t SB_OFF = S_OFF  + (size_t)MTOK * DD * 4;      // slots bf16
  constexpr size_t FB_OFF = SB_OFF + (size_t)MTOK * DD * 2;      // [f|b] concat bf16
  constexpr size_t H_OFF  = FB_OFF + (size_t)MTOK * DINNER * 2;  // hidden bf16
  constexpr size_t W_OFF  = H_OFF  + (size_t)MTOK * DFF * 2;     // bf16 weights
  constexpr size_t W_ELEMS =
      (size_t)NLAYER * (2 * DD * DINNER +        // fw_w1, bw_w1
                        2 * DINNER * DD +        // fw_w2, bw_w2
                        2 * DD * DD +            // mg_w (512x256)
                        DD * DFF + DFF * DD);    // ffn_w1, ffn_w2
  constexpr size_t NEED = W_OFF + W_ELEMS * 2;
  if (ws_size < NEED) return;

  char* ws = (char*)d_ws;
  float* S  = (float*)(ws + S_OFF);
  bf16*  Sb = (bf16*)(ws + SB_OFF);
  bf16*  FB = (bf16*)(ws + FB_OFF);
  bf16*  H  = (bf16*)(ws + H_OFF);
  bf16*  Wb = (bf16*)(ws + W_OFF);

  bf16* fw_w1b  = Wb;
  bf16* fw_w2b  = fw_w1b  + (size_t)NLAYER * DD * DINNER;
  bf16* bw_w1b  = fw_w2b  + (size_t)NLAYER * DINNER * DD;
  bf16* bw_w2b  = bw_w1b  + (size_t)NLAYER * DD * DINNER;
  bf16* mg_wb   = bw_w2b  + (size_t)NLAYER * DINNER * DD;
  bf16* ffn_w1b = mg_wb   + (size_t)NLAYER * 2 * DD * DD;
  bf16* ffn_w2b = ffn_w1b + (size_t)NLAYER * DD * DFF;

  auto cvt = [&](const float* s, bf16* d, int n) {
    cvt_f32_bf16<<<dim3((n + 255) / 256), dim3(256), 0, stream>>>(s, d, n);
  };
  cvt(fw_w1,  fw_w1b,  NLAYER * DD * DINNER);
  cvt(fw_w2,  fw_w2b,  NLAYER * DINNER * DD);
  cvt(bw_w1,  bw_w1b,  NLAYER * DD * DINNER);
  cvt(bw_w2,  bw_w2b,  NLAYER * DINNER * DD);
  cvt(mg_w,   mg_wb,   NLAYER * 2 * DD * DD);
  cvt(ffn_w1, ffn_w1b, NLAYER * DD * DFF);
  cvt(ffn_w2, ffn_w2b, NLAYER * DFF * DD);
  cvt(slots,  Sb,      MTOK * DD);
  hipMemcpyAsync(S, slots, (size_t)MTOK * DD * 4, hipMemcpyDeviceToDevice, stream);

  const dim3 blk(256);
  const int GX = MTOK / 32;  // 4096 row-tiles

  for (int i = 0; i < NLAYER; ++i) {
    // fw branch: H = SiLU(Sb @ fw_w1 + b1); FB[:, :256] = LN(H @ fw_w2 + b2)
    wmma_mlp<EPI_SILU><<<dim3(GX, DINNER / 256), blk, 0, stream>>>(
        Sb, DD, fw_w1b + (size_t)i * DD * DINNER, DINNER, fw_b1 + i * DINNER, DD,
        H, DINNER, nullptr, nullptr, nullptr, nullptr, nullptr, 0, nullptr, 0, nullptr);
    wmma_mlp<EPI_LN><<<dim3(GX, 1), blk, 0, stream>>>(
        H, DINNER, fw_w2b + (size_t)i * DINNER * DD, DD, fw_b2 + i * DD, DINNER,
        FB, DINNER, fw_lnw + i * DD, fw_lnb + i * DD, nullptr, nullptr,
        nullptr, 0, nullptr, 0, nullptr);
    // bw branch: FB[:, 256:] = LN(SiLU(Sb @ bw_w1 + b1) @ bw_w2 + b2)
    wmma_mlp<EPI_SILU><<<dim3(GX, DINNER / 256), blk, 0, stream>>>(
        Sb, DD, bw_w1b + (size_t)i * DD * DINNER, DINNER, bw_b1 + i * DINNER, DD,
        H, DINNER, nullptr, nullptr, nullptr, nullptr, nullptr, 0, nullptr, 0, nullptr);
    wmma_mlp<EPI_LN><<<dim3(GX, 1), blk, 0, stream>>>(
        H, DINNER, bw_w2b + (size_t)i * DINNER * DD, DD, bw_b2 + i * DD, DINNER,
        FB + DD, DINNER, bw_lnw + i * DD, bw_lnb + i * DD, nullptr, nullptr,
        nullptr, 0, nullptr, 0, nullptr);
    // merge: S = LN_nm(S + FB @ mg_w + mg_b); Sb = LN_ffn(S)
    wmma_mlp<EPI_MERGE><<<dim3(GX, 1), blk, 0, stream>>>(
        FB, DINNER, mg_wb + (size_t)i * 2 * DD * DD, DD, mg_b + i * DD, DINNER,
        Sb, DD, nm_w + i * DD, nm_b + i * DD, ffn_lnw + i * DD, ffn_lnb + i * DD,
        S, DD, S, DD, nullptr);
    // FFN: H = GELU(Sb @ ffn_w1 + b1); out = (S + H @ ffn_w2 + b2) * keep
    wmma_mlp<EPI_GELU><<<dim3(GX, DFF / 256), blk, 0, stream>>>(
        Sb, DD, ffn_w1b + (size_t)i * DD * DFF, DFF, ffn_b1 + i * DFF, DD,
        H, DFF, nullptr, nullptr, nullptr, nullptr, nullptr, 0, nullptr, 0, nullptr);
    float* outS = (i == NLAYER - 1) ? (float*)d_out : S;
    wmma_mlp<EPI_OUT><<<dim3(GX, 1), blk, 0, stream>>>(
        H, DFF, ffn_w2b + (size_t)i * DFF * DD, DD, ffn_b2 + i * DD, DFF,
        Sb, DD, nullptr, nullptr, nullptr, nullptr, S, DD, outS, DD, keep);
  }
}
